// MeshGraphDecoder_81535659148050
// MI455X (gfx1250) — compile-verified, matrix operand order checked by reading, not run on hardware
//
#include <hip/hip_runtime.h>
#include <stdint.h>
#include <stddef.h>

// ---------- CDNA5 wave32 WMMA types ----------
typedef __attribute__((ext_vector_type(16))) __bf16 v16bf;
typedef __attribute__((ext_vector_type(8)))  __bf16 v8bf;
typedef __attribute__((ext_vector_type(8)))  float  v8f;

#define DD     512
#define NGRID  65536
#define EEDGE  196608
#define TM     32      // rows (edges / grid nodes) per workgroup

union FragU { v16bf v; v8bf h[2]; };

// Load one 16x32 bf16 WMMA fragment (A-style layout; B uses same layout when
// the weight matrix is stored transposed [N][K]).
// lane l<16: row=lr, K = k0+khalf+{0..7} and k0+khalf+16+{0..7}; khalf=8 for l>=16.
__device__ __forceinline__ v16bf load_frag(const __bf16* p0, int ld, int lr,
                                           int khalf, int k0) {
  FragU u;
  const __bf16* p = p0 + (size_t)lr * ld + (size_t)(k0 + khalf);
  u.h[0] = *(const v8bf*)(p);
  u.h[1] = *(const v8bf*)(p + 16);
  return u.v;
}

// One wave computes a 32(M) x 64(N) tile: 2 M-tiles x 4 N-tiles of 16x16.
// A: LDS bf16 [32][ldA]; WT: global bf16 [N][K] (transposed weights).
template <int KSTEPS>
__device__ __forceinline__ void wave_gemm(const __bf16* A, int ldA,
                                          const __bf16* WT, int K,
                                          int lr, int khalf, int n0,
                                          v8f acc[2][4]) {
#pragma unroll 2
  for (int ks = 0; ks < KSTEPS; ++ks) {
    const int k0 = ks * 32;
    v16bf b0 = load_frag(WT + (size_t)(n0 +  0) * K, K, lr, khalf, k0);
    v16bf b1 = load_frag(WT + (size_t)(n0 + 16) * K, K, lr, khalf, k0);
    v16bf b2 = load_frag(WT + (size_t)(n0 + 32) * K, K, lr, khalf, k0);
    v16bf b3 = load_frag(WT + (size_t)(n0 + 48) * K, K, lr, khalf, k0);
    v16bf a0 = load_frag(A,                    ldA, lr, khalf, k0);
    v16bf a1 = load_frag(A + (size_t)16 * ldA, ldA, lr, khalf, k0);
    acc[0][0] = __builtin_amdgcn_wmma_f32_16x16x32_bf16(false, a0, false, b0, (short)0, acc[0][0], false, false);
    acc[0][1] = __builtin_amdgcn_wmma_f32_16x16x32_bf16(false, a0, false, b1, (short)0, acc[0][1], false, false);
    acc[0][2] = __builtin_amdgcn_wmma_f32_16x16x32_bf16(false, a0, false, b2, (short)0, acc[0][2], false, false);
    acc[0][3] = __builtin_amdgcn_wmma_f32_16x16x32_bf16(false, a0, false, b3, (short)0, acc[0][3], false, false);
    acc[1][0] = __builtin_amdgcn_wmma_f32_16x16x32_bf16(false, a1, false, b0, (short)0, acc[1][0], false, false);
    acc[1][1] = __builtin_amdgcn_wmma_f32_16x16x32_bf16(false, a1, false, b1, (short)0, acc[1][1], false, false);
    acc[1][2] = __builtin_amdgcn_wmma_f32_16x16x32_bf16(false, a1, false, b2, (short)0, acc[1][2], false, false);
    acc[1][3] = __builtin_amdgcn_wmma_f32_16x16x32_bf16(false, a1, false, b3, (short)0, acc[1][3], false, false);
  }
}

// ---------- weight convert + transpose: fp32 [K][512] -> bf16 [512][K] ----------
__global__ void wconv_kernel(const float* __restrict__ in, __bf16* __restrict__ out, int K) {
  int idx = blockIdx.x * blockDim.x + threadIdx.x;
  int total = K * DD;
  if (idx >= total) return;
  int n = idx / K, k = idx - n * K;           // writes coalesced along k
  out[(size_t)n * K + k] = (__bf16)in[(size_t)k * DD + n];
}

// ---------- edge MLP + scatter-add ----------
__global__ __launch_bounds__(256, 2)
void edge_mlp_scatter(const float* __restrict__ efeat,
                      const float* __restrict__ gridf,
                      const float* __restrict__ meshf,
                      const int* __restrict__ src_idx,
                      const int* __restrict__ dst_idx,
                      const __bf16* __restrict__ wt0, const float* __restrict__ bias0,
                      const __bf16* __restrict__ wt1, const float* __restrict__ bias1,
                      const float* __restrict__ lng, const float* __restrict__ lnb,
                      float* agg) {
  extern __shared__ char smem[];
  const int ldA = 1536 + 8;                 // 3088B row stride -> 4 dword skew, no bank conflicts
  const int ldH = DD + 8;                   // 1040B row stride
  __bf16* Abuf = (__bf16*)smem;             // [32][1544] bf16 = 98816B
  __bf16* Hbuf = (__bf16*)smem;             // [32][520]  bf16 (reuses Abuf region)
  __shared__ int   dsts[TM], srcs[TM];
  __shared__ float red[TM][8][2];
  __shared__ float stats[TM][2];

  const int tid = threadIdx.x;
  const int e0  = blockIdx.x * TM;
  if (tid < TM)            dsts[tid]      = dst_idx[e0 + tid];
  else if (tid < 2 * TM)   srcs[tid - TM] = src_idx[e0 + tid - TM];
  __syncthreads();

  // Gather concat(efeat, mesh[src], grid[dst]) = [32][1536] fp32 -> bf16 LDS
  for (int it = tid; it < TM * 384; it += 256) {
    int r = it / 384, q = it - r * 384;
    int seg = q >> 7, off = (q & 127) << 2;
    const float* sp;
    if (seg == 0)      sp = efeat + (size_t)(e0 + r) * DD + off;
    else if (seg == 1) sp = meshf + (size_t)srcs[r] * DD + off;
    else               sp = gridf + (size_t)dsts[r] * DD + off;
    float4 v = *(const float4*)sp;
    __bf16* dp = Abuf + (size_t)r * ldA + seg * DD + off;
    dp[0] = (__bf16)v.x; dp[1] = (__bf16)v.y; dp[2] = (__bf16)v.z; dp[3] = (__bf16)v.w;
  }
  __syncthreads();

  const int wave  = tid >> 5, lane = tid & 31;
  const int lr    = lane & 15;
  const int khalf = (lane >> 4) << 3;       // 0 or 8 (K and M offset of upper half)
  const int n0    = wave * 64;

  // GEMM1: [32,1536] x [1536,512]
  v8f acc[2][4] = {};
  wave_gemm<48>(Abuf, ldA, wt0, 1536, lr, khalf, n0, acc);
  __syncthreads();

  // SiLU(acc + b0) -> bf16 LDS
#pragma unroll
  for (int ni = 0; ni < 4; ++ni) {
    const int col = n0 + ni * 16 + lr;
    const float bb = bias0[col];
#pragma unroll
    for (int mi = 0; mi < 2; ++mi)
#pragma unroll
      for (int i = 0; i < 8; ++i) {
        const int row = mi * 16 + khalf + i;
        float x = acc[mi][ni][i] + bb;
        float h = x / (1.0f + __expf(-x));
        Hbuf[(size_t)row * ldH + col] = (__bf16)h;
      }
  }
  __syncthreads();

  // GEMM2: [32,512] x [512,512]
  v8f acc2[2][4] = {};
  wave_gemm<16>(Hbuf, ldH, wt1, DD, lr, khalf, n0, acc2);

  float b1v[4], gv[4], bv[4];
#pragma unroll
  for (int ni = 0; ni < 4; ++ni) {
    const int col = n0 + ni * 16 + lr;
    b1v[ni] = bias1[col]; gv[ni] = lng[col]; bv[ni] = lnb[col];
  }

  // LayerNorm stats straight from registers: 16-lane butterfly then cross-wave LDS
#pragma unroll
  for (int mi = 0; mi < 2; ++mi)
#pragma unroll
    for (int i = 0; i < 8; ++i) {
      float s = 0.f, s2 = 0.f;
#pragma unroll
      for (int ni = 0; ni < 4; ++ni) {
        float v = acc2[mi][ni][i] + b1v[ni];
        s += v; s2 += v * v;
      }
#pragma unroll
      for (int m = 8; m >= 1; m >>= 1) {
        s  += __shfl_xor(s,  m);
        s2 += __shfl_xor(s2, m);
      }
      if (lr == 0) {
        const int row = mi * 16 + khalf + i;
        red[row][wave][0] = s;
        red[row][wave][1] = s2;
      }
    }
  __syncthreads();
  if (tid < TM) {
    float s = 0.f, s2 = 0.f;
#pragma unroll
    for (int w = 0; w < 8; ++w) { s += red[tid][w][0]; s2 += red[tid][w][1]; }
    const float mu  = s * (1.0f / DD);
    const float var = s2 * (1.0f / DD) - mu * mu;
    stats[tid][0] = mu;
    stats[tid][1] = rsqrtf(var + 1e-5f);
  }
  __syncthreads();

  // Normalize + scatter-add into agg (f32 global atomics)
#pragma unroll
  for (int mi = 0; mi < 2; ++mi)
#pragma unroll
    for (int i = 0; i < 8; ++i) {
      const int row = mi * 16 + khalf + i;
      const float mu = stats[row][0], rs = stats[row][1];
      float* op = agg + (size_t)dsts[row] * DD;
#pragma unroll
      for (int ni = 0; ni < 4; ++ni) {
        const int col = n0 + ni * 16 + lr;
        float v = acc2[mi][ni][i] + b1v[ni];
        atomicAdd(op + col, (v - mu) * rs * gv[ni] + bv[ni]);
      }
    }
}

// ---------- node MLP + residual (agg read from d_out, final written to d_out) ----------
__global__ __launch_bounds__(256, 2)
void node_mlp(const float* agg,             // == out (read fully before overwrite)
              const float* __restrict__ gridf,
              const __bf16* __restrict__ wt0, const float* __restrict__ bias0,
              const __bf16* __restrict__ wt1, const float* __restrict__ bias1,
              const float* __restrict__ lng, const float* __restrict__ lnb,
              float* out) {
  extern __shared__ char smem[];
  const int ldA = 1024 + 8;                 // 2064B stride -> 4 dword skew
  const int ldH = DD + 8;
  __bf16* Abuf = (__bf16*)smem;             // [32][1032] bf16 = 66048B
  __bf16* Hbuf = (__bf16*)smem;
  __shared__ float red[TM][8][2];
  __shared__ float stats[TM][2];

  const int tid = threadIdx.x;
  const int r0  = blockIdx.x * TM;

  // Gather concat(agg, grid_nfeat) = [32][1024] fp32 -> bf16 LDS
  for (int it = tid; it < TM * 256; it += 256) {
    int r = it >> 8, q = it & 255;
    int seg = q >> 7, off = (q & 127) << 2;
    const float* sp = (seg == 0) ? agg   + (size_t)(r0 + r) * DD + off
                                 : gridf + (size_t)(r0 + r) * DD + off;
    float4 v = *(const float4*)sp;
    __bf16* dp = Abuf + (size_t)r * ldA + seg * DD + off;
    dp[0] = (__bf16)v.x; dp[1] = (__bf16)v.y; dp[2] = (__bf16)v.z; dp[3] = (__bf16)v.w;
  }
  __syncthreads();

  const int wave  = tid >> 5, lane = tid & 31;
  const int lr    = lane & 15;
  const int khalf = (lane >> 4) << 3;
  const int n0    = wave * 64;

  v8f acc[2][4] = {};
  wave_gemm<32>(Abuf, ldA, wt0, 1024, lr, khalf, n0, acc);
  __syncthreads();

#pragma unroll
  for (int ni = 0; ni < 4; ++ni) {
    const int col = n0 + ni * 16 + lr;
    const float bb = bias0[col];
#pragma unroll
    for (int mi = 0; mi < 2; ++mi)
#pragma unroll
      for (int i = 0; i < 8; ++i) {
        const int row = mi * 16 + khalf + i;
        float x = acc[mi][ni][i] + bb;
        float h = x / (1.0f + __expf(-x));
        Hbuf[(size_t)row * ldH + col] = (__bf16)h;
      }
  }
  __syncthreads();

  v8f acc2[2][4] = {};
  wave_gemm<16>(Hbuf, ldH, wt1, DD, lr, khalf, n0, acc2);

  float b1v[4], gv[4], bv[4];
#pragma unroll
  for (int ni = 0; ni < 4; ++ni) {
    const int col = n0 + ni * 16 + lr;
    b1v[ni] = bias1[col]; gv[ni] = lng[col]; bv[ni] = lnb[col];
  }

#pragma unroll
  for (int mi = 0; mi < 2; ++mi)
#pragma unroll
    for (int i = 0; i < 8; ++i) {
      float s = 0.f, s2 = 0.f;
#pragma unroll
      for (int ni = 0; ni < 4; ++ni) {
        float v = acc2[mi][ni][i] + b1v[ni];
        s += v; s2 += v * v;
      }
#pragma unroll
      for (int m = 8; m >= 1; m >>= 1) {
        s  += __shfl_xor(s,  m);
        s2 += __shfl_xor(s2, m);
      }
      if (lr == 0) {
        const int row = mi * 16 + khalf + i;
        red[row][wave][0] = s;
        red[row][wave][1] = s2;
      }
    }
  __syncthreads();
  if (tid < TM) {
    float s = 0.f, s2 = 0.f;
#pragma unroll
    for (int w = 0; w < 8; ++w) { s += red[tid][w][0]; s2 += red[tid][w][1]; }
    const float mu  = s * (1.0f / DD);
    const float var = s2 * (1.0f / DD) - mu * mu;
    stats[tid][0] = mu;
    stats[tid][1] = rsqrtf(var + 1e-5f);
  }
  __syncthreads();

  // LN + residual, plain stores (rows disjoint per block; agg fully consumed above)
#pragma unroll
  for (int mi = 0; mi < 2; ++mi)
#pragma unroll
    for (int i = 0; i < 8; ++i) {
      const int row = mi * 16 + khalf + i;
      const float mu = stats[row][0], rs = stats[row][1];
      const size_t rb = (size_t)(r0 + row) * DD;
#pragma unroll
      for (int ni = 0; ni < 4; ++ni) {
        const int col = n0 + ni * 16 + lr;
        float v = acc2[mi][ni][i] + b1v[ni];
        out[rb + col] = (v - mu) * rs * gv[ni] + bv[ni] + gridf[rb + col];
      }
    }
}

extern "C" void kernel_launch(void* const* d_in, const int* in_sizes, int n_in,
                              void* d_out, int out_size, void* d_ws, size_t ws_size,
                              hipStream_t stream) {
  const float* m2g    = (const float*)d_in[0];
  const float* gridf  = (const float*)d_in[1];
  const float* meshf  = (const float*)d_in[2];
  const int*   srcidx = (const int*)d_in[3];
  const int*   dstidx = (const int*)d_in[4];
  const float* e_w0 = (const float*)d_in[5];
  const float* e_b0 = (const float*)d_in[6];
  const float* e_w1 = (const float*)d_in[7];
  const float* e_b1 = (const float*)d_in[8];
  const float* e_lg = (const float*)d_in[9];
  const float* e_lb = (const float*)d_in[10];
  const float* n_w0 = (const float*)d_in[11];
  const float* n_b0 = (const float*)d_in[12];
  const float* n_w1 = (const float*)d_in[13];
  const float* n_b1 = (const float*)d_in[14];
  const float* n_lg = (const float*)d_in[15];
  const float* n_lb = (const float*)d_in[16];

  // Workspace: bf16 transposed weights (~3.7 MB total)
  __bf16* wt_e0 = (__bf16*)d_ws;                 // [512][1536]
  __bf16* wt_e1 = wt_e0 + (size_t)512 * 1536;    // [512][512]
  __bf16* wt_n0 = wt_e1 + (size_t)512 * 512;     // [512][1024]
  __bf16* wt_n1 = wt_n0 + (size_t)512 * 1024;    // [512][512]

  wconv_kernel<<<(1536 * 512) / 256, 256, 0, stream>>>(e_w0, wt_e0, 1536);
  wconv_kernel<<<(512  * 512) / 256, 256, 0, stream>>>(e_w1, wt_e1, 512);
  wconv_kernel<<<(1024 * 512) / 256, 256, 0, stream>>>(n_w0, wt_n0, 1024);
  wconv_kernel<<<(512  * 512) / 256, 256, 0, stream>>>(n_w1, wt_n1, 512);

  // d_out doubles as the scatter accumulator; zero it every call (capture-safe).
  hipMemsetAsync(d_out, 0, (size_t)NGRID * DD * sizeof(float), stream);

  edge_mlp_scatter<<<EEDGE / TM, 256, 32 * (1536 + 8) * sizeof(__bf16), stream>>>(
      m2g, gridf, meshf, srcidx, dstidx,
      wt_e0, e_b0, wt_e1, e_b1, e_lg, e_lb, (float*)d_out);

  node_mlp<<<NGRID / TM, 256, 32 * (1024 + 8) * sizeof(__bf16), stream>>>(
      (const float*)d_out, gridf,
      wt_n0, n_b0, wt_n1, n_b1, n_lg, n_lb, (float*)d_out);
}
